// DiscreteComms_13005160973101
// MI455X (gfx1250) — matrix operand order repaired and checked by reading, not compiled, not run on hardware
//
#include <hip/hip_runtime.h>
#include <hip/hip_bf16.h>
#include <math.h>

typedef __attribute__((ext_vector_type(16))) __bf16 v16bf;
typedef __attribute__((ext_vector_type(8)))  __bf16 v8bf;
typedef __attribute__((ext_vector_type(4)))  __bf16 v4bf;
typedef __attribute__((ext_vector_type(8)))  float  v8f;
typedef __attribute__((ext_vector_type(4)))  unsigned int u32x4;
typedef __attribute__((ext_vector_type(8)))  int  i32x8;
typedef __attribute__((ext_vector_type(4)))  int  i32x4;

#define M_ROWS 8192   // B*T*N
#define H_DIM  512
#define O_DIM  512    // NC*CD
#define NC_    8
#define CD_    64
#define V_     1024
#define R_ROWS 65536  // M_ROWS * NC

// LDS strides (in halves) produced by TDM padding
#define WROW_ST 520   // 512 + 8 halves pad  -> 1040 B rows (16B aligned, bank-spread)
#define CROW_ST 72    // 64 + 8 halves pad   -> 144 B rows  (16B aligned, bank-spread)

// ---------------------------------------------------------------------------
// Tensor Data Mover: 2D bf16 tile global -> LDS with per-row LDS padding.
// clang-23 / amdgpu-toolchain 6-arg builtin form:
//   (uint32x4 g0, int32x8 g1, int32x4 g2, int32x4 g3, int32x8, i32 cpol)
// pad_interval: 2^(n+1) dwords, pad_amount: n+1 dwords.
// ---------------------------------------------------------------------------
__device__ __forceinline__ void tdm_load_2d_bf16(
    unsigned int lds_off, const void* gaddr,
    unsigned int dim0, unsigned int dim1,
    unsigned int tile0, unsigned int tile1,
    unsigned int stride0,
    unsigned int pad_interval, unsigned int pad_amount) {
  unsigned long long ga = (unsigned long long)(size_t)gaddr;
  u32x4 g0;
  g0[0] = 1u;                                          // count=1 (user D#)
  g0[1] = lds_off;                                     // LDS byte address
  g0[2] = (unsigned int)(ga & 0xFFFFFFFFu);            // global addr [31:0]
  g0[3] = (unsigned int)((ga >> 32) & 0x1FFFFFFu)      // global addr [56:32]
          | (2u << 30);                                // type = 2 (image)
  i32x8 g1;
  g1[0] = (int)((1u << 16)                             // data_size = 2 bytes
          | (1u << 20)                                 // pad_enable
          | (pad_interval << 22) | (pad_amount << 25));
  g1[1] = (int)((dim0 & 0xFFFFu) << 16);               // tensor_dim0[15:0]
  g1[2] = (int)(((dim0 >> 16) & 0xFFFFu) | ((dim1 & 0xFFFFu) << 16));
  g1[3] = (int)(((dim1 >> 16) & 0xFFFFu) | ((tile0 & 0xFFFFu) << 16));
  g1[4] = (int)(tile1 & 0xFFFFu);                      // tile_dim1; tile_dim2=0
  g1[5] = (int)stride0;                                // tensor_dim0_stride lo
  g1[6] = 0;
  g1[7] = 0;
  i32x4 z4 = {};
  i32x8 z8 = {};
  __builtin_amdgcn_tensor_load_to_lds(g0, g1, z4, z4, z8, 0);
}

// ---------------------------------------------------------------------------
// Kernel 0: f32 -> bf16 conversion (x, W, codebook), one b64 store per thread
// ---------------------------------------------------------------------------
__global__ __launch_bounds__(256)
void cvt_bf16_kernel(const float* __restrict__ src, __bf16* __restrict__ dst, int n4) {
  const int i = blockIdx.x * 256 + threadIdx.x;
  if (i < n4) {
    float4 f = ((const float4*)src)[i];
    v4bf o;
    o[0] = (__bf16)f.x; o[1] = (__bf16)f.y; o[2] = (__bf16)f.z; o[3] = (__bf16)f.w;
    *(v4bf*)(dst + 4 * (size_t)i) = o;
  }
}

// ---------------------------------------------------------------------------
// Kernel 1: latent = tanh(xbf @ Wbf^T + b) as bf16 [65536 x 64].
// Block = 256 thr (8 waves); block tile 128(M) x 128(O).
// Wbf tile (128 rows x 512) staged by one TDM into padded LDS.
// Each wave: 16(M) x 128(O) strip.  Per k-step: 1 A fragment, then all 8 B
// fragments preloaded into distinct regs, then an 8-WMMA chain (no WAR nops).
// ---------------------------------------------------------------------------
__global__ __launch_bounds__(256)
void proj_tanh_kernel(const __bf16* __restrict__ xbf, const __bf16* __restrict__ Wbf,
                      const float* __restrict__ bias, __bf16* __restrict__ latent) {
  extern __shared__ char smem[];
  __bf16* sW = (__bf16*)smem;                 // [128][WROW_ST]

  const int tid = threadIdx.x;
  const int o0  = blockIdx.y * 128;

  if (tid < 32) {
    // one TDM per block: 128 rows x 512 halves, pad 4 dwords per 256 dwords (1 row)
    tdm_load_2d_bf16((unsigned int)(size_t)sW, Wbf + (size_t)o0 * H_DIM,
                     H_DIM, O_DIM, H_DIM, 128, H_DIM,
                     /*pad_interval=*/7, /*pad_amount=*/3);
    __builtin_amdgcn_s_wait_tensorcnt(0);
  }
  __syncthreads();

  const int wave = tid >> 5, lane = tid & 31;
  const int hi   = lane >> 4;
  const int kb   = hi * 8;
  const int m0   = blockIdx.x * 128 + wave * 16;
  const int col  = lane & 15;
  const __bf16* xrow = xbf + (size_t)(m0 + col) * H_DIM;

  // Two hoisted LDS row bases so all ds_loads use immediate offsets (< 64KB):
  // rb0 covers W-tile rows 0..63 (s = 0..3), rb1 rows 64..127 (s = 4..7).
  const __bf16* rb0 = sW + (size_t)col * WROW_ST + hi * 8;
  const __bf16* rb1 = rb0 + (size_t)64 * WROW_ST;

  v8f acc[8];
  for (int s = 0; s < 8; ++s) acc[s] = (v8f){};

  for (int k0 = 0; k0 < H_DIM; k0 += 32) {
    // A fragment (bf16, no conversion): K pattern {kb..kb+7, kb+16..kb+23}
    v8bf p0 = *(const v8bf*)(xrow + k0 + kb);
    v8bf p1 = *(const v8bf*)(xrow + k0 + kb + 16);
    v16bf a;
    for (int i = 0; i < 8; ++i) { a[i] = p0[i]; a[i + 8] = p1[i]; }

    // Preload all 8 B fragments, then issue the WMMA chain back-to-back.
    v16bf bm[8];
    for (int s = 0; s < 4; ++s) {
      const __bf16* wr = rb0 + (size_t)s * 16 * WROW_ST + k0;
      v8bf q0 = *(const v8bf*)wr;
      v8bf q1 = *(const v8bf*)(wr + 16);
      for (int i = 0; i < 8; ++i) { bm[s][i] = q0[i]; bm[s][i + 8] = q1[i]; }
    }
    for (int s = 0; s < 4; ++s) {
      const __bf16* wr = rb1 + (size_t)s * 16 * WROW_ST + k0;
      v8bf q0 = *(const v8bf*)wr;
      v8bf q1 = *(const v8bf*)(wr + 16);
      for (int i = 0; i < 8; ++i) { bm[s + 4][i] = q0[i]; bm[s + 4][i + 8] = q1[i]; }
    }
    for (int s = 0; s < 8; ++s) {
      acc[s] = __builtin_amdgcn_wmma_f32_16x16x32_bf16(
          false, a, false, bm[s], (short)0, acc[s], false, false);
    }
  }

  for (int s = 0; s < 8; ++s) {
    const int   o  = o0 + s * 16 + col;
    const float bv = bias[o];
    for (int j = 0; j < 8; ++j) {
      const int rm = m0 + j + hi * 8;
      latent[(size_t)rm * O_DIM + o] = (__bf16)tanhf(acc[s][j] + bv);
    }
  }
}

// ---------------------------------------------------------------------------
// Kernel 2: row norms of latent (bf16) and codebook (f32)
// ---------------------------------------------------------------------------
__global__ __launch_bounds__(256)
void norms_kernel(const __bf16* __restrict__ latent, const float* __restrict__ codebook,
                  float* __restrict__ anorm, float* __restrict__ cnorm) {
  const int t = blockIdx.x * 256 + threadIdx.x;
  if (t < R_ROWS) {
    const __bf16* row = latent + (size_t)t * CD_;
    float s = 0.f;
    for (int i = 0; i < CD_; ++i) { float v = (float)row[i]; s += v * v; }
    anorm[t] = s;
  } else if (t < R_ROWS + V_) {
    const int r = t - R_ROWS;
    const float* row = codebook + (size_t)r * CD_;
    float s = 0.f;
    for (int i = 0; i < CD_; ++i) s += row[i] * row[i];
    cnorm[r] = s;
  }
}

// ---------------------------------------------------------------------------
// Kernel 3: -sqrt(d2) logits + fused online log-softmax / entropy / selected
// log-prob.  Codebook (bf16) staged once per block by TDM into padded LDS.
// ---------------------------------------------------------------------------
__global__ __launch_bounds__(256)
void dist_softmax_kernel(const __bf16* __restrict__ latent,
                         const __bf16* __restrict__ cbf,
                         const int*   __restrict__ comms,
                         const float* __restrict__ anorm,
                         const float* __restrict__ cnorm,
                         float* __restrict__ out_lp, float* __restrict__ out_ent) {
  extern __shared__ char dsm[];
  __bf16* cbS  = (__bf16*)dsm;                          // [1024][CROW_ST]
  float*  slog = (float*)(dsm + (size_t)V_ * CROW_ST * 2); // [8][16][17]

  const int tid  = threadIdx.x;
  if (tid < 32) {
    // 1024 rows x 64 halves, pad 4 dwords per 32 dwords (1 row) -> stride 72 halves
    tdm_load_2d_bf16((unsigned int)(size_t)cbS, cbf,
                     CD_, V_, CD_, V_, CD_,
                     /*pad_interval=*/4, /*pad_amount=*/3);
    __builtin_amdgcn_s_wait_tensorcnt(0);
  }
  __syncthreads();

  const int wave = tid >> 5, lane = tid & 31;
  const int hi   = lane >> 4;
  const int kb   = hi * 8;
  const int r0   = blockIdx.x * 128 + wave * 16;
  const int mr   = lane & 15;
  const __bf16* lrow = latent + (size_t)(r0 + mr) * CD_;

  // Two A fragments (k-steps 0, 32), loaded once, reused for all 64 tiles.
  v16bf aA, aB;
  {
    v8bf p0 = *(const v8bf*)(lrow + kb);
    v8bf p1 = *(const v8bf*)(lrow + kb + 16);
    v8bf p2 = *(const v8bf*)(lrow + 32 + kb);
    v8bf p3 = *(const v8bf*)(lrow + 32 + kb + 16);
    for (int i = 0; i < 8; ++i) {
      aA[i] = p0[i]; aA[i + 8] = p1[i];
      aB[i] = p2[i]; aB[i + 8] = p3[i];
    }
  }
  float4 an0 = *(const float4*)(anorm + r0 + hi * 8);
  float4 an1 = *(const float4*)(anorm + r0 + hi * 8 + 4);
  const float anv[8] = {an0.x, an0.y, an0.z, an0.w, an1.x, an1.y, an1.z, an1.w};
  const int selidx = comms[r0 + mr];

  float mrun = -INFINITY, S = 0.f, T = 0.f, lsel = 0.f;
  float* myrow = slog + (wave * 16 + mr) * 17;

  for (int c0 = 0; c0 < V_; c0 += 16) {
    const int cv = c0 + mr;
    const __bf16* crow = cbS + (size_t)cv * CROW_ST;
    v16bf b0, b1;
    {
      v8bf g0 = *(const v8bf*)(crow + hi * 8);
      v8bf g1 = *(const v8bf*)(crow + hi * 8 + 16);
      v8bf h0 = *(const v8bf*)(crow + 32 + hi * 8);
      v8bf h1 = *(const v8bf*)(crow + 32 + hi * 8 + 16);
      for (int i = 0; i < 8; ++i) {
        b0[i] = g0[i]; b0[i + 8] = g1[i];
        b1[i] = h0[i]; b1[i + 8] = h1[i];
      }
    }
    v8f acc = {};
    acc = __builtin_amdgcn_wmma_f32_16x16x32_bf16(
        false, aA, false, b0, (short)0, acc, false, false);
    acc = __builtin_amdgcn_wmma_f32_16x16x32_bf16(
        false, aB, false, b1, (short)0, acc, false, false);

    const float cnv = cnorm[cv];
    for (int j = 0; j < 8; ++j) {
      float d2 = anv[j] + cnv - 2.f * acc[j];
      slog[(wave * 16 + j + hi * 8) * 17 + mr] = -sqrtf(fmaxf(d2, 0.f));
    }
    __syncthreads();

    float lv[16], tmax = -INFINITY;
    for (int i = 0; i < 16; ++i) { lv[i] = myrow[i]; tmax = fmaxf(tmax, lv[i]); }
    float st = 0.f, tt = 0.f;
    for (int i = 0; i < 16; ++i) { float e = expf(lv[i] - tmax); st += e; tt += lv[i] * e; }
    const float nm = fmaxf(mrun, tmax);
    const float s1 = expf(mrun - nm), s2 = expf(tmax - nm);
    S = S * s1 + st * s2;
    T = T * s1 + tt * s2;
    mrun = nm;
    if (selidx >= c0 && selidx < c0 + 16) lsel = lv[selidx - c0];
    __syncthreads();
  }

  const float logZ = mrun + logf(S);
  float ent = logZ - T / S;
  float lp  = lsel - logZ;
  for (int off = 1; off < 8; off <<= 1) {
    lp  += __shfl_xor(lp,  off);
    ent += __shfl_xor(ent, off);
  }
  if (lane < 16 && (lane & 7) == 0) {
    const int m = (r0 + lane) >> 3;
    out_lp[m]  = lp;
    out_ent[m] = ent;
  }
}

// ---------------------------------------------------------------------------
// Kernel 4: forward straight-through output = codebook[comms] (f32 gather)
// ---------------------------------------------------------------------------
__global__ __launch_bounds__(256)
void gather_out_kernel(const int* __restrict__ comms,
                       const float* __restrict__ codebook,
                       float* __restrict__ out0) {
  const int g = blockIdx.x * 256 + threadIdx.x;   // float4 index, 1M total
  const int r = g >> 4;
  const int q = g & 15;
  const int v = comms[r];
  ((float4*)out0)[g] = ((const float4*)codebook)[v * 16 + q];
}

// ---------------------------------------------------------------------------
extern "C" void kernel_launch(void* const* d_in, const int* in_sizes, int n_in,
                              void* d_out, int out_size, void* d_ws, size_t ws_size,
                              hipStream_t stream) {
  (void)in_sizes; (void)n_in; (void)out_size; (void)ws_size;
  const float* x        = (const float*)d_in[0];
  const int*   comms    = (const int*)  d_in[1];
  const float* W        = (const float*)d_in[2];
  const float* bias     = (const float*)d_in[3];
  const float* codebook = (const float*)d_in[4];

  float* out     = (float*)d_out;
  float* out0    = out;                               // [8192 x 512]
  float* out_lp  = out + (size_t)M_ROWS * O_DIM;      // [8192]
  float* out_ent = out_lp + M_ROWS;                   // [8192]

  char*   ws     = (char*)d_ws;
  __bf16* xb     = (__bf16*)ws;                                   // 8 MB
  __bf16* Wb     = xb + (size_t)M_ROWS * H_DIM;                   // 512 KB
  __bf16* cb     = Wb + (size_t)O_DIM * H_DIM;                    // 128 KB
  __bf16* latent = cb + (size_t)V_ * CD_;                         // 8 MB
  float*  anorm  = (float*)(latent + (size_t)R_ROWS * CD_);       // 256 KB
  float*  cnorm  = anorm + R_ROWS;                                // 4 KB

  const size_t lds1 = 128 * WROW_ST * 2;                  // 133,120 B
  const size_t lds3 = (size_t)V_ * CROW_ST * 2 + 8 * 16 * 17 * 4; // 156,160 B
  (void)hipFuncSetAttribute((const void*)proj_tanh_kernel,
                            hipFuncAttributeMaxDynamicSharedMemorySize, (int)lds1);
  (void)hipFuncSetAttribute((const void*)dist_softmax_kernel,
                            hipFuncAttributeMaxDynamicSharedMemorySize, (int)lds3);

  cvt_bf16_kernel<<<(M_ROWS * H_DIM / 4 + 255) / 256, 256, 0, stream>>>(x, xb, M_ROWS * H_DIM / 4);
  cvt_bf16_kernel<<<(O_DIM * H_DIM / 4 + 255) / 256, 256, 0, stream>>>(W, Wb, O_DIM * H_DIM / 4);
  cvt_bf16_kernel<<<(V_ * CD_ / 4 + 255) / 256, 256, 0, stream>>>(codebook, cb, V_ * CD_ / 4);

  dim3 g1(M_ROWS / 128, O_DIM / 128, 1);
  proj_tanh_kernel<<<g1, 256, lds1, stream>>>(xb, Wb, bias, latent);
  norms_kernel<<<(R_ROWS + V_ + 255) / 256, 256, 0, stream>>>(latent, codebook, anorm, cnorm);
  dist_softmax_kernel<<<R_ROWS / 128, 256, lds3, stream>>>(latent, cb, comms,
                                                           anorm, cnorm, out_lp, out_ent);
  gather_out_kernel<<<(M_ROWS * O_DIM / 4) / 256, 256, 0, stream>>>(comms, codebook, out0);
}
